// GCNStoModel_MultiHead_21483426415084
// MI455X (gfx1250) — matrix operand order repaired and checked by reading, not compiled, not run on hardware
//
#include <hip/hip_runtime.h>
#include <hip/hip_bf16.h>

// ---------------- types ----------------
typedef __attribute__((ext_vector_type(16))) __bf16 v16bf;
typedef __attribute__((ext_vector_type(8)))  __bf16 v8bf;
typedef __attribute__((ext_vector_type(8)))  float  v8f;
typedef __attribute__((ext_vector_type(4)))  float  v4f;

// ---------------- problem constants (from reference) ----------------
#define C_IN    128
#define C_HID   256
#define C_OUT   256
#define C_PHEAD 64
#define C_N0    1048576
#define C_E0    2097152
#define C_ND0   131072
#define C_E1    131072
#define C_ND1   8192

static __device__ __forceinline__ void atom_add_f32(float* p, float v) {
  // Relaxed agent-scope fadd -> native global_atomic_add_f32 (no return, STOREcnt)
  __hip_atomic_fetch_add(p, v, __ATOMIC_RELAXED, __HIP_MEMORY_SCOPE_AGENT);
}

// ---------------- zero ----------------
__global__ void k_zero_f32(float* __restrict__ p, long long n4) {
  long long i = (long long)blockIdx.x * blockDim.x + threadIdx.x;
  if (i < n4) {
    v4f z = {};
    ((v4f*)p)[i] = z;
  }
}

// ---------------- f32 -> bf16 convert ----------------
__global__ void k_cvt_bf16(const float* __restrict__ src, __bf16* __restrict__ dst, int n) {
  int i = blockIdx.x * blockDim.x + threadIdx.x;
  if (i < n) dst[i] = (__bf16)src[i];
}

// ---------------- layer-0 edge scatter: f32 feats, FEAT=128, one wave per edge ----------------
// Each lane loads 4 consecutive floats (one b128) and issues 4 f32 atomics.
__global__ void k_scatter_f32(const float* __restrict__ feat, const int* __restrict__ src,
                              const int* __restrict__ dst, float* __restrict__ neigh,
                              float* __restrict__ deg, int nedge) {
  int wave = (int)(((long long)blockIdx.x * blockDim.x + threadIdx.x) >> 5);
  int lane = threadIdx.x & 31;
  if (wave >= nedge) return;
  int s = src[wave];
  int d = dst[wave];
  const v4f* fs = (const v4f*)(feat + (size_t)s * C_IN);
  float* nd = neigh + (size_t)d * C_IN + lane * 4;
  v4f v = fs[lane];
#pragma unroll
  for (int j = 0; j < 4; ++j) atom_add_f32(nd + j, v[j]);
  if (lane == 0) atom_add_f32(deg + d, 1.0f);
}

// ---------------- layer-1 edge scatter: bf16 feats, FEAT=256, one wave per edge ----------------
// Each lane loads 8 consecutive bf16 (one b128), converts, and issues 8 f32 atomics.
__global__ void k_scatter_bf16(const __bf16* __restrict__ feat, const int* __restrict__ src,
                               const int* __restrict__ dst, float* __restrict__ neigh,
                               float* __restrict__ deg, int nedge) {
  int wave = (int)(((long long)blockIdx.x * blockDim.x + threadIdx.x) >> 5);
  int lane = threadIdx.x & 31;
  if (wave >= nedge) return;
  int s = src[wave];
  int d = dst[wave];
  const v8bf* fs = (const v8bf*)(feat + (size_t)s * C_HID);
  float* nd = neigh + (size_t)d * C_HID + lane * 8;
  v8bf v = fs[lane];
#pragma unroll
  for (int j = 0; j < 8; ++j) atom_add_f32(nd + j, (float)v[j]);
  if (lane == 0) atom_add_f32(deg + d, 1.0f);
}

// ---------------- normalize: (neigh + dstfeat) / (deg + 1) -> bf16 ----------------
__global__ void k_norm_f32dst(const float* __restrict__ neigh, const float* __restrict__ xdst,
                              const float* __restrict__ deg, __bf16* __restrict__ out,
                              long long total, int featshift) {
  long long i = (long long)blockIdx.x * blockDim.x + threadIdx.x;
  if (i >= total) return;
  long long m = i >> featshift;
  float r = (neigh[i] + xdst[i]) / (deg[m] + 1.0f);
  out[i] = (__bf16)r;
}

__global__ void k_norm_bf16dst(const float* __restrict__ neigh, const __bf16* __restrict__ hdst,
                               const float* __restrict__ deg, __bf16* __restrict__ out,
                               long long total, int featshift) {
  long long i = (long long)blockIdx.x * blockDim.x + threadIdx.x;
  if (i >= total) return;
  long long m = i >> featshift;
  float r = (neigh[i] + (float)hdst[i]) / (deg[m] + 1.0f);
  out[i] = (__bf16)r;
}

// ---------------- bf16 WMMA GEMM:  Out[M,N] = relu?(A[M,K] * W[N,K]^T + bias[N]) ----------------
// One wave computes a 16x64 output tile: a single A fragment (16x32) feeds 4
// v_wmma_f32_16x16x32_bf16 against 4 B fragments, stepping K by 32. This cuts
// A-matrix L2 re-reads 4x vs a 16x16 tile and gives 4 independent WMMAs per load group.
// Fragment layouts used here (A and B share one contiguous K order):
//   A 16x32 (MxK):  lane m=l&15; lanes 0-15 hold K 0..15; lanes 16-31 hold K 16..31
//   B 32x16 (KxN):  lane n=l&15; lanes 0-15 hold K 0..15; lanes 16-31 hold K 16..31
//   C/D 16x16 f32:  lane n=l&15; VGPR v -> row (v + 8*(l>>4))
// Requires N % 64 == 0 and M % 16 == 0; grids are exact so EXEC is all-1s at every WMMA.
__global__ void k_gemm_bf16(const __bf16* __restrict__ A, const __bf16* __restrict__ W,
                            const float* __restrict__ bias, float* __restrict__ outF,
                            __bf16* __restrict__ outB, int M, int N, int K, int do_relu) {
  int gw   = (int)(((long long)blockIdx.x * blockDim.x + threadIdx.x) >> 5);
  int lane = threadIdx.x & 31;
  int ntile64 = N >> 6;
  int tm = gw / ntile64;
  int tn = gw - tm * ntile64;        // which group of 64 output columns
  if (tm * 16 >= M) return;          // uniform per wave -> EXEC stays all-1s

  int l15 = lane & 15;
  int hf  = lane >> 4;

  const __bf16* arow = A + (size_t)(tm * 16 + l15) * K;
  int nbase = tn * 64 + l15;
  const __bf16* w0 = W + (size_t)(nbase +  0) * K + hf * 16;
  const __bf16* w1 = W + (size_t)(nbase + 16) * K + hf * 16;
  const __bf16* w2 = W + (size_t)(nbase + 32) * K + hf * 16;
  const __bf16* w3 = W + (size_t)(nbase + 48) * K + hf * 16;

  v8f acc0 = {}, acc1 = {}, acc2 = {}, acc3 = {};
  for (int kb = 0; kb < K; kb += 32) {
    v8bf alo = *(const v8bf*)(arow + kb + hf * 16);       // K = kb + hf*16 .. +7
    v8bf ahi = *(const v8bf*)(arow + kb + 8 + hf * 16);   // K = kb + 8 + hf*16 .. +7
    v16bf a = __builtin_shufflevector(alo, ahi,
                                      0, 1, 2, 3, 4, 5, 6, 7,
                                      8, 9, 10, 11, 12, 13, 14, 15);
    v16bf b0 = *(const v16bf*)(w0 + kb);
    v16bf b1 = *(const v16bf*)(w1 + kb);
    v16bf b2 = *(const v16bf*)(w2 + kb);
    v16bf b3 = *(const v16bf*)(w3 + kb);
    acc0 = __builtin_amdgcn_wmma_f32_16x16x32_bf16(false, a, false, b0, (short)0, acc0, false, false);
    acc1 = __builtin_amdgcn_wmma_f32_16x16x32_bf16(false, a, false, b1, (short)0, acc1, false, false);
    acc2 = __builtin_amdgcn_wmma_f32_16x16x32_bf16(false, a, false, b2, (short)0, acc2, false, false);
    acc3 = __builtin_amdgcn_wmma_f32_16x16x32_bf16(false, a, false, b3, (short)0, acc3, false, false);
  }

  int mbase = tm * 16 + hf * 8;
  v8f accs[4] = {acc0, acc1, acc2, acc3};
#pragma unroll
  for (int j = 0; j < 4; ++j) {
    int n = nbase + j * 16;
    float bv = bias[n];
#pragma unroll
    for (int v = 0; v < 8; ++v) {
      float r = accs[j][v] + bv;
      if (do_relu) r = fmaxf(r, 0.0f);
      size_t o = (size_t)(mbase + v) * (size_t)N + (size_t)n;
      if (outB) outB[o] = (__bf16)r;
      else      outF[o] = r;
    }
  }
}

// ---------------- host launch ----------------
extern "C" void kernel_launch(void* const* d_in, const int* in_sizes, int n_in,
                              void* d_out, int out_size, void* d_ws, size_t ws_size,
                              hipStream_t stream) {
  (void)in_sizes; (void)n_in; (void)out_size; (void)ws_size;

  const float* x    = (const float*)d_in[0];
  const int*   src0 = (const int*)  d_in[1];
  const int*   dst0 = (const int*)  d_in[2];
  const int*   src1 = (const int*)  d_in[3];
  const int*   dst1 = (const int*)  d_in[4];
  const float* W1   = (const float*)d_in[5];
  const float* b1   = (const float*)d_in[6];
  const float* W2   = (const float*)d_in[7];
  const float* b2   = (const float*)d_in[8];
  const float* Wh   = (const float*)d_in[9];
  const float* bh   = (const float*)d_in[10];
  float* out = (float*)d_out;

  // ---- workspace layout (all offsets multiples of 256B) ----
  char* ws = (char*)d_ws;
  size_t off = 0;
  float* neigh0 = (float*)(ws + off); off += (size_t)C_ND0 * C_IN  * 4;  // 64 MB
  float* deg0   = (float*)(ws + off); off += (size_t)C_ND0 * 4;          // 0.5 MB
  float* neigh1 = (float*)(ws + off); off += (size_t)C_ND1 * C_HID * 4;  // 8 MB
  float* deg1   = (float*)(ws + off); off += (size_t)C_ND1 * 4;          // 32 KB
  const long long zero_f32s = (long long)C_ND0 * C_IN + C_ND0 + (long long)C_ND1 * C_HID + C_ND1;
  __bf16* a0bf = (__bf16*)(ws + off); off += (size_t)C_ND0 * C_IN  * 2;  // 32 MB
  __bf16* hbf  = (__bf16*)(ws + off); off += (size_t)C_ND0 * C_HID * 2;  // 64 MB
  __bf16* a1bf = (__bf16*)(ws + off); off += (size_t)C_ND1 * C_HID * 2;  // 4 MB
  __bf16* h2bf = (__bf16*)(ws + off); off += (size_t)C_ND1 * C_HID * 2;  // 4 MB
  __bf16* W1bf = (__bf16*)(ws + off); off += (size_t)C_HID * C_IN   * 2;
  __bf16* W2bf = (__bf16*)(ws + off); off += (size_t)C_OUT * C_HID  * 2;
  __bf16* Whbf = (__bf16*)(ws + off); off += (size_t)C_PHEAD * C_OUT * 2;

  // 1) zero the contiguous accumulator block (neigh0|deg0|neigh1|deg1)
  {
    long long n4 = zero_f32s / 4;
    int blocks = (int)((n4 + 255) / 256);
    k_zero_f32<<<blocks, 256, 0, stream>>>(neigh0, n4);
  }

  // 2) convert weights to bf16
  k_cvt_bf16<<<(C_HID * C_IN    + 255) / 256, 256, 0, stream>>>(W1, W1bf, C_HID * C_IN);
  k_cvt_bf16<<<(C_OUT * C_HID   + 255) / 256, 256, 0, stream>>>(W2, W2bf, C_OUT * C_HID);
  k_cvt_bf16<<<(C_PHEAD * C_OUT + 255) / 256, 256, 0, stream>>>(Wh, Whbf, C_PHEAD * C_OUT);

  // 3) layer-0 scatter: wave per edge (8 edges per 256-thread block)
  k_scatter_f32<<<C_E0 / 8, 256, 0, stream>>>(x, src0, dst0, neigh0, deg0, C_E0);

  // 4) normalize: (neigh0 + x[:ND0]) / (deg0 + 1) -> bf16 A0
  {
    long long total = (long long)C_ND0 * C_IN;
    k_norm_f32dst<<<(int)((total + 255) / 256), 256, 0, stream>>>(neigh0, x, deg0, a0bf, total, 7);
  }

  // 5) GEMM1: h = relu(A0[131072x128] * W1[256x128]^T + b1) -> bf16
  {
    int waves = (C_ND0 / 16) * (C_HID / 64);           // 32768 waves (16x64 tiles)
    k_gemm_bf16<<<waves * 32 / 256, 256, 0, stream>>>(a0bf, W1bf, b1, nullptr, hbf,
                                                      C_ND0, C_HID, C_IN, 1);
  }

  // 6) layer-1 scatter: wave per edge over 256 bf16 feats
  k_scatter_bf16<<<C_E1 / 8, 256, 0, stream>>>(hbf, src1, dst1, neigh1, deg1, C_E1);

  // 7) normalize: (neigh1 + h[:ND1]) / (deg1 + 1) -> bf16 A1
  {
    long long total = (long long)C_ND1 * C_HID;
    k_norm_bf16dst<<<(int)((total + 255) / 256), 256, 0, stream>>>(neigh1, hbf, deg1, a1bf, total, 8);
  }

  // 8) GEMM2: h2 = relu(A1[8192x256] * W2[256x256]^T + b2) -> bf16
  {
    int waves = (C_ND1 / 16) * (C_OUT / 64);           // 2048 waves
    k_gemm_bf16<<<waves * 32 / 256, 256, 0, stream>>>(a1bf, W2bf, b2, nullptr, h2bf,
                                                      C_ND1, C_OUT, C_HID, 1);
  }

  // 9) head: out = h2[8192x256] * Wh[64x256]^T + bh -> f32 d_out
  {
    int waves = (C_ND1 / 16) * (C_PHEAD / 64);         // 512 waves
    k_gemm_bf16<<<waves * 32 / 256, 256, 0, stream>>>(h2bf, Whbf, bh, out, nullptr,
                                                      C_ND1, C_PHEAD, C_OUT, 0);
  }
}
